// ReciprocalLayer3_82257213653387
// MI455X (gfx1250) — compile-verified, matrix operand order checked by reading, not run on hardware
//
#include <hip/hip_runtime.h>

// ---------------------------------------------------------------------------
// MI455X / gfx1250 implementation of the GPS + reciprocal-attention model.
// All matmuls (GEMMs, QK^T, PV) run on the matrix units via
// v_wmma_f32_16x16x32_f16 (f16 operands, f32 accumulation). Both GEMM
// operands are pre-packed to f16 once per GEMM (W additionally transposed),
// so the hot loop is purely contiguous 16B loads + WMMA -- zero inner-loop
// conversions. Attention is flash-style streaming (no 8192x8192 scores
// materialized) with Q/K packed f16 and V packed f16-transposed.
// wave32: 256-thread blocks = 8 waves; GEMM waves own a 64x16 tile.
// ---------------------------------------------------------------------------

typedef _Float16 v16h __attribute__((ext_vector_type(16)));
typedef float    v8f  __attribute__((ext_vector_type(8)));

__device__ __forceinline__ v8f wmma16x16x32(v16h a, v16h b, v8f c) {
  return __builtin_amdgcn_wmma_f32_16x16x32_f16(
      /*neg_a=*/false, a, /*neg_b=*/false, b,
      /*c_mod=*/(short)0, c, /*reuse_a=*/false, /*reuse_b=*/false);
}

// f16 fragment: lane L picks row (row0 + L&15), reads contiguous K halves.
// Per ISA 7.12.2 (16-bit A 16x32): elements 0..7 -> K = k0 + half*8 + i,
// 8..15 -> +16, half = L>>4. Serves GEMM A (packed f16), GEMM B (packed
// W^T), flash Q, flash K^T, flash V^T -- all contiguous 16B loads.
__device__ __forceinline__ v16h load_frag_f16(const _Float16* __restrict__ X,
                                              int ld, int row0, int k0, int lane) {
  const _Float16* p = X + (long)(row0 + (lane & 15)) * ld + k0 + ((lane >> 4) << 3);
  v16h r;
#pragma unroll
  for (int i = 0; i < 8; ++i) {
    r[i]     = p[i];
    r[i + 8] = p[i + 16];
  }
  return r;
}

// ---------------------------------------------------------------------------
// Packing kernels (one linear pass, L2 resident; amortized over the N/16
// column-blocks that re-read each operand).
// ---------------------------------------------------------------------------
// out[c*R + r] = (f16) in[r*ld + c0 + c]  (transpose-pack: W^T / V^T)
__global__ void packT_kernel(const float* __restrict__ in, int R, int C, int ld, int c0,
                             _Float16* __restrict__ out) {
  long i = (long)blockIdx.x * blockDim.x + threadIdx.x;
  if (i >= (long)R * C) return;
  int c = (int)(i / R), r = (int)(i % R);
  out[(long)c * R + r] = (_Float16)in[(long)r * ld + c0 + c];
}
// out[r*C + c] = (f16) in[r*ld + c0 + c]  (straight f16 pack of a col-slice)
__global__ void pack_kernel(const float* __restrict__ in, int R, int C, int ld, int c0,
                            _Float16* __restrict__ out) {
  long i = (long)blockIdx.x * blockDim.x + threadIdx.x;
  if (i >= (long)R * C) return;
  int r = (int)(i / C), c = (int)(i % C);
  out[i] = (_Float16)in[(long)r * ld + c0 + c];
}

// ---------------------------------------------------------------------------
// GEMM: C[M,N] = A[M,K] @ W[K,N] (+bias) (+residual) (relu?).
// A pre-packed f16 [M,K]; W pre-packed f16-transposed [N,K].
// M % 512 == 0, N % 16 == 0, K % 32 == 0 (true for every call here).
// grid = (N/16, M/512), block = 256 (8 waves), one 64x16 tile per wave:
// one B fragment feeds 4 WMMAs; all operand loads are contiguous b128s.
// ---------------------------------------------------------------------------
template <bool HAS_BIAS, bool HAS_RES, bool RELU>
__global__ void gemm_kernel(const _Float16* __restrict__ A, const _Float16* __restrict__ WT,
                            const float* __restrict__ bias, const float* __restrict__ res,
                            float* __restrict__ C, int M, int N, int K) {
  const int lane = threadIdx.x & 31;
  const int wv   = threadIdx.x >> 5;
  const int row0 = (blockIdx.y * 8 + wv) * 64;
  const int col0 = blockIdx.x * 16;

  v8f acc[4];
#pragma unroll
  for (int t = 0; t < 4; ++t) acc[t] = (v8f){0.f, 0.f, 0.f, 0.f, 0.f, 0.f, 0.f, 0.f};

  for (int k0 = 0; k0 < K; k0 += 32) {
    // gfx1250 global_prefetch_b8: speculative, unconditional (no branch).
    __builtin_prefetch(A + (long)(row0 + (lane & 15)) * K + k0 + 32, 0, 1);
    v16h a0 = load_frag_f16(A, K, row0,      k0, lane);
    v16h a1 = load_frag_f16(A, K, row0 + 16, k0, lane);
    v16h a2 = load_frag_f16(A, K, row0 + 32, k0, lane);
    v16h a3 = load_frag_f16(A, K, row0 + 48, k0, lane);
    v16h b  = load_frag_f16(WT, K, col0, k0, lane);
    acc[0] = wmma16x16x32(a0, b, acc[0]);
    acc[1] = wmma16x16x32(a1, b, acc[1]);
    acc[2] = wmma16x16x32(a2, b, acc[2]);
    acc[3] = wmma16x16x32(a3, b, acc[3]);
  }

  const int n  = col0 + (lane & 15);
  const float bv = HAS_BIAS ? bias[n] : 0.f;
#pragma unroll
  for (int t = 0; t < 4; ++t) {
    const int mbase = row0 + t * 16 + ((lane >> 4) << 3);  // M = r + 8*half
#pragma unroll
    for (int r = 0; r < 8; ++r) {
      long idx = (long)(mbase + r) * N + n;
      float v = acc[t][r] + bv;
      if (HAS_RES) v += res[idx];
      if (RELU) v = fmaxf(v, 0.f);
      C[idx] = v;
    }
  }
}

// ---------------------------------------------------------------------------
// Flash attention: one 16-row query tile per wave, 32 keys per iteration.
// Q [Lq, H*DH] f16, K [Lk, H*DH] f16, V^T [H*DH, Lk] f16. Heads on grid.y.
// DH = head dim = value dim (64 recip, 128 GPS). Output f32, head h at
// column h*ohs of O (row stride ldo).
// ---------------------------------------------------------------------------
template <int DH>
__global__ void flash_kernel(const _Float16* __restrict__ Q, int ldq,
                             const _Float16* __restrict__ Km, int ldk,
                             const _Float16* __restrict__ VT, int ldvt,
                             float* __restrict__ O, int ldo, int ohs,
                             int Lq, int Lk, float scale) {
  __shared__ _Float16 pbuf[8 * 16 * 32];  // 8 KB: one 16x32 P tile per wave
  const int lane = threadIdx.x & 31;
  const int wv   = threadIdx.x >> 5;
  _Float16* P    = pbuf + wv * (16 * 32);

  const int h  = blockIdx.y;
  const int q0 = blockIdx.x * 128 + wv * 16;
  const _Float16* qh  = Q  + (long)h * DH;
  const _Float16* kh  = Km + (long)h * DH;
  const _Float16* vth = VT + (long)h * DH * ldvt;
  float*          oh  = O  + (long)h * ohs;

  v16h qf[DH / 32];
#pragma unroll
  for (int c = 0; c < DH / 32; ++c) qf[c] = load_frag_f16(qh, ldq, q0, c * 32, lane);

  v8f of[DH / 16];
#pragma unroll
  for (int c = 0; c < DH / 16; ++c) of[c] = (v8f){0.f, 0.f, 0.f, 0.f, 0.f, 0.f, 0.f, 0.f};
  float rm[8], rs[8], scf[8];
#pragma unroll
  for (int r = 0; r < 8; ++r) { rm[r] = -1e30f; rs[r] = 0.f; }

  for (int key0 = 0; key0 < Lk; key0 += 32) {
    // S tiles: s0 = keys [key0, key0+16), s1 = keys [key0+16, key0+32)
    v8f s0 = {0.f, 0.f, 0.f, 0.f, 0.f, 0.f, 0.f, 0.f};
    v8f s1 = {0.f, 0.f, 0.f, 0.f, 0.f, 0.f, 0.f, 0.f};
#pragma unroll
    for (int c = 0; c < DH / 32; ++c) {
      // B = K^T: lane's column n = key row -> contiguous f16 load.
      s0 = wmma16x16x32(qf[c], load_frag_f16(kh, ldk, key0,      c * 32, lane), s0);
      s1 = wmma16x16x32(qf[c], load_frag_f16(kh, ldk, key0 + 16, c * 32, lane), s1);
    }
    // Online softmax per query row (row M = r + 8*half lives across the
    // 16 lanes of this half-wave; xor masks < 16 stay inside the half).
#pragma unroll
    for (int r = 0; r < 8; ++r) {
      float a0 = s0[r] * scale, a1 = s1[r] * scale;
      float mx = fmaxf(a0, a1);
#pragma unroll
      for (int off = 1; off < 16; off <<= 1) mx = fmaxf(mx, __shfl_xor(mx, off, 32));
      float nm = fmaxf(rm[r], mx);
      scf[r]   = __expf(rm[r] - nm);
      rm[r]    = nm;
      float p0 = __expf(a0 - nm), p1 = __expf(a1 - nm);
      s0[r] = p0; s1[r] = p1;
      float sum = p0 + p1;
#pragma unroll
      for (int off = 1; off < 16; off <<= 1) sum += __shfl_xor(sum, off, 32);
      rs[r] = rs[r] * scf[r] + sum;
    }
#pragma unroll
    for (int c = 0; c < DH / 16; ++c)
#pragma unroll
      for (int r = 0; r < 8; ++r) of[c][r] *= scf[r];
    // Stage P (16x32 f16) through per-wave LDS to re-layout C-frag -> A-frag.
#pragma unroll
    for (int r = 0; r < 8; ++r) {
      int m = r + ((lane >> 4) << 3);
      P[m * 32 + (lane & 15)]      = (_Float16)s0[r];
      P[m * 32 + 16 + (lane & 15)] = (_Float16)s1[r];
    }
    v16h pa = load_frag_f16(P, 32, 0, 0, lane);
#pragma unroll
    for (int c = 0; c < DH / 16; ++c)
      of[c] = wmma16x16x32(pa, load_frag_f16(vth, ldvt, c * 16, key0, lane), of[c]);
  }

  const int n  = lane & 15;
  const int mb = q0 + ((lane >> 4) << 3);
#pragma unroll
  for (int c = 0; c < DH / 16; ++c)
#pragma unroll
    for (int r = 0; r < 8; ++r)
      oh[(long)(mb + r) * ldo + c * 16 + n] = of[c][r] / rs[r];
}

// ---------------------------------------------------------------------------
// GIN aggregation: agg[dst] += node[src] (atomic scatter over edges).
// ---------------------------------------------------------------------------
__global__ void segsum_kernel(const float* __restrict__ node, const int* __restrict__ ei,
                              float* __restrict__ agg, long total, int D) {
  long i = (long)blockIdx.x * blockDim.x + threadIdx.x;
  if (i >= total) return;
  int e = (int)(i / D), d = (int)(i % D);
  int src = ei[2 * e], dst = ei[2 * e + 1];
  atomicAdd(&agg[(long)dst * D + d], node[(long)src * D + d]);
}

__global__ void add2_kernel(const float* __restrict__ a, const float* __restrict__ b,
                            float* __restrict__ y, long n) {
  long i = (long)blockIdx.x * blockDim.x + threadIdx.x;
  if (i < n) y[i] = a[i] + b[i];
}

// BatchNorm stats (training-mode, ddof=0), row-chunk parallel:
// grid = (C/128, Nr/512); partial sums accumulated with atomics.
__global__ void bn_partial_kernel(const float* __restrict__ x, int Nr, int C,
                                  float* __restrict__ acc) {
  int c = blockIdx.x * blockDim.x + threadIdx.x;
  if (c >= C) return;
  const int chunk = Nr / gridDim.y;
  const int r0 = blockIdx.y * chunk;
  float s1 = 0.f, s2 = 0.f;
  for (int r = r0; r < r0 + chunk; ++r) {
    float v = x[(long)r * C + c];
    s1 += v; s2 += v * v;
  }
  atomicAdd(&acc[c], s1);
  atomicAdd(&acc[C + c], s2);
}

__global__ void bn_finalize_kernel(const float* __restrict__ acc, int Nr, int C,
                                   float* __restrict__ mu, float* __restrict__ inv) {
  int c = blockIdx.x * blockDim.x + threadIdx.x;
  if (c >= C) return;
  float m = acc[c] / Nr;
  mu[c]  = m;
  inv[c] = rsqrtf(acc[C + c] / Nr - m * m + 1e-5f);
}

__global__ void bn_apply_kernel(const float* __restrict__ x, const float* __restrict__ mu,
                                const float* __restrict__ inv, const float* __restrict__ g,
                                const float* __restrict__ b, float* __restrict__ y,
                                long total, int C, int relu) {
  long i = (long)blockIdx.x * blockDim.x + threadIdx.x;
  if (i >= total) return;
  int c = (int)(i % C);
  float v = g[c] * (x[i] - mu[c]) * inv[c] + b[c];
  if (relu) v = fmaxf(v, 0.f);
  y[i] = v;
}

// LayerNorm over last dim of (x + res), eps 1e-5. One block per row.
__global__ void ln_res_kernel(const float* __restrict__ x, const float* __restrict__ res,
                              const float* __restrict__ g, const float* __restrict__ b,
                              float* __restrict__ y, int D) {
  __shared__ float sred[2][8];
  const long base = (long)blockIdx.x * D;
  float s1 = 0.f, s2 = 0.f;
  for (int i = threadIdx.x; i < D; i += blockDim.x) {
    float v = x[base + i] + res[base + i];
    s1 += v; s2 += v * v;
  }
#pragma unroll
  for (int off = 16; off; off >>= 1) { s1 += __shfl_xor(s1, off, 32); s2 += __shfl_xor(s2, off, 32); }
  int wv = threadIdx.x >> 5, lane = threadIdx.x & 31;
  if (lane == 0) { sred[0][wv] = s1; sred[1][wv] = s2; }
  __syncthreads();
  int nw = blockDim.x >> 5;
  if (wv == 0) {
    s1 = lane < nw ? sred[0][lane] : 0.f;
    s2 = lane < nw ? sred[1][lane] : 0.f;
#pragma unroll
    for (int off = 16; off; off >>= 1) { s1 += __shfl_xor(s1, off, 32); s2 += __shfl_xor(s2, off, 32); }
    if (lane == 0) { sred[0][0] = s1; sred[1][0] = s2; }
  }
  __syncthreads();
  float mu  = sred[0][0] / D;
  float inv = rsqrtf(sred[1][0] / D - mu * mu + 1e-5f);
  for (int i = threadIdx.x; i < D; i += blockDim.x) {
    float v = x[base + i] + res[base + i];
    y[base + i] = g[i] * (v - mu) * inv + b[i];
  }
}

// ---------------------------------------------------------------------------
// Host-side orchestration
// ---------------------------------------------------------------------------
namespace {

constexpr long NP = 8192, NQ = 1024, DD = 384, HIDG = 768;

inline unsigned cdiv256(long n) { return (unsigned)((n + 255) / 256); }

inline void gemm(const float* A, const float* W, const float* bias, const float* res,
                 float* C, int M, int N, int K, int relu,
                 _Float16* wpack, _Float16* apack, hipStream_t s) {
  packT_kernel<<<cdiv256((long)K * N), 256, 0, s>>>(W, K, N, N, 0, wpack);  // W^T f16 [N,K]
  pack_kernel <<<cdiv256((long)M * K), 256, 0, s>>>(A, M, K, K, 0, apack);  // A f16 [M,K]
  dim3 g(N / 16, M / 512);
  if (res)       gemm_kernel<true,  true,  false><<<g, 256, 0, s>>>(apack, wpack, bias, res, C, M, N, K);
  else if (relu) gemm_kernel<true,  false, true ><<<g, 256, 0, s>>>(apack, wpack, bias, res, C, M, N, K);
  else if (bias) gemm_kernel<true,  false, false><<<g, 256, 0, s>>>(apack, wpack, bias, res, C, M, N, K);
  else           gemm_kernel<false, false, false><<<g, 256, 0, s>>>(apack, wpack, bias, res, C, M, N, K);
}
inline void bn(const float* x, float* y, const float* g, const float* b,
               int Nr, int C, int relu, float* acc, float* mu, float* inv, hipStream_t s) {
  hipMemsetAsync(acc, 0, 2 * C * sizeof(float), s);
  bn_partial_kernel<<<dim3((C + 127) / 128, Nr / 512), 128, 0, s>>>(x, Nr, C, acc);
  bn_finalize_kernel<<<(C + 127) / 128, 128, 0, s>>>(acc, Nr, C, mu, inv);
  long total = (long)Nr * C;
  bn_apply_kernel<<<cdiv256(total), 256, 0, s>>>(x, mu, inv, g, b, y, total, C, relu);
}
inline void add2(const float* a, const float* b, float* y, long n, hipStream_t s) {
  add2_kernel<<<cdiv256(n), 256, 0, s>>>(a, b, y, n);
}

// One full GPS layer. P = 24 parameter pointers in _make_gps_params order.
void run_gps(const float* X, const int* ei, long E, int N, const float* const* P,
             float* gout, float* node, float* ba, float* bb, float* bc, float* big,
             _Float16* fQ, _Float16* fK, _Float16* fV, _Float16* wpack, _Float16* apack,
             float* acc, float* mu, float* inv, hipStream_t s) {
  const long ND = (long)N * DD;
  // node = X @ emb_w + emb_b
  gemm(X, P[0], P[1], nullptr, node, N, DD, DD, 0, wpack, apack, s);
  // GIN: agg = segment_sum(node[src] -> dst); g = node + agg
  hipMemsetAsync(ba, 0, ND * sizeof(float), s);
  long tot = E * DD;
  segsum_kernel<<<cdiv256(tot), 256, 0, s>>>(node, ei, ba, tot, DD);
  add2(node, ba, ba, ND, s);
  gemm(ba, P[2], P[3], nullptr, big, N, HIDG, DD, 0, wpack, apack, s);         // g@gin_w1+b1
  bn(big, big, P[4], P[5], N, HIDG, /*relu=*/1, acc, mu, inv, s);              // relu(bn(.))
  gemm(big, P[6], P[7], node /*+node*/, bb, N, DD, HIDG, 0, wpack, apack, s);  // @gin_w2+b2+node
  bn(bb, bb, P[12], P[13], N, DD, 0, acc, mu, inv, s);                         // h1
  // Attention branch: qkv = node @ attn_in + b -> [N,1152]; pack Q/K f16, V^T f16
  gemm(node, P[8], P[9], nullptr, big, N, 1152, DD, 0, wpack, apack, s);
  pack_kernel <<<cdiv256(ND), 256, 0, s>>>(big, N, DD, 1152, 0,   fQ);
  pack_kernel <<<cdiv256(ND), 256, 0, s>>>(big, N, DD, 1152, 384, fK);
  packT_kernel<<<cdiv256(ND), 256, 0, s>>>(big, N, DD, 1152, 768, fV);         // V^T [384,N]
  flash_kernel<128><<<dim3(N / 128, 3), 256, 0, s>>>(
      fQ, 384, fK, 384, fV, N, ba, 384, 128, N, N, 1.0f / sqrtf(128.0f));      // attn -> ba
  gemm(ba, P[10], P[11], node /*+node*/, bc, N, DD, DD, 0, wpack, apack, s);   // @attn_out+b+node
  bn(bc, bc, P[14], P[15], N, DD, 0, acc, mu, inv, s);                         // h2
  add2(bb, bc, ba, ND, s);                                                     // out = h1+h2
  gemm(ba, P[16], P[17], nullptr, big, N, HIDG, DD, 1, wpack, apack, s);       // relu(out@mlp_w1+b1)
  gemm(big, P[18], P[19], ba /*+out*/, bc, N, DD, HIDG, 0, wpack, apack, s);   // @mlp_w2+b2 + out
  bn(bc, bc, P[20], P[21], N, DD, 0, acc, mu, inv, s);                         // n3
  add2(bc, node, bc, ND, s);                                                   // + node (GPS residual)
  bn(bc, gout, P[22], P[23], N, DD, 0, acc, mu, inv, s);                       // outer bn
}

}  // namespace

extern "C" void kernel_launch(void* const* d_in, const int* in_sizes, int n_in,
                              void* d_out, int out_size, void* d_ws, size_t ws_size,
                              hipStream_t stream) {
  auto f = [&](int i) { return (const float*)d_in[i]; };
  const float* pep_enc  = f(0);   // [1,1024,384]
  const float* prot_enc = f(1);   // [1,8192,384]
  const int* pep_ei  = (const int*)d_in[2];   // [1,32768,2]
  const int* prot_ei = (const int*)d_in[3];   // [1,262144,2]
  const long EQ = in_sizes[2] / 2, EP = in_sizes[3] / 2;

  // params flattened in reference insertion order
  const float* Pp[24]; for (int i = 0; i < 24; ++i) Pp[i] = f(4 + i);    // prot_gps
  const float* Pq[24]; for (int i = 0; i < 24; ++i) Pq[i] = f(28 + i);   // pep_gps
  const float* wq = f(52), *wk = f(53), *wv = f(54), *wv2 = f(55);
  const float* fc1_w = f(56), *fc1_b = f(57), *fc2_w = f(58), *fc2_b = f(59);
  const float* ln1_g = f(60), *ln1_b = f(61), *ln2_g = f(62), *ln2_b = f(63);
  const float* fpw1 = f(64), *fpb1 = f(65), *fpw2 = f(66), *fpb2 = f(67), *fpg = f(68), *fpb = f(69);
  const float* fqw1 = f(70), *fqb1 = f(71), *fqw2 = f(72), *fqb2 = f(73), *fqg = f(74), *fqb = f(75);

  // Workspace arenas (~180 MB total)
  float* w = (float*)d_ws;
  float* node  = w; w += NP * DD;
  float* ba    = w; w += NP * DD;
  float* bb    = w; w += NP * DD;
  float* bc    = w; w += NP * DD;
  float* big   = w; w += NP * 1536;
  float* goutp = w; w += NP * DD;
  float* goutq = w; w += NQ * DD;
  float* accb  = w; w += 4096;
  float* mu    = w; w += 2048;
  float* inv   = w; w += 2048;
  _Float16* fh = (_Float16*)w;           // f16 arena
  _Float16* fQ    = fh;                  // <= NP*512 halves
  _Float16* fK    = fh + NP * 512;       // <= NP*512
  _Float16* fV    = fh + 2 * NP * 512;   // <= NP*512
  _Float16* wpack = fh + 3 * NP * 512;   // <= 1M halves (max K*N = 384*1536)
  _Float16* apack = wpack + 1024 * 1024; // <= NP*1536 halves (max M*K)

  // --- GPS layers ---
  run_gps(prot_enc, prot_ei, EP, (int)NP, Pp, goutp, node, ba, bb, bc, big,
          fQ, fK, fV, wpack, apack, accb, mu, inv, stream);
  run_gps(pep_enc,  pep_ei,  EQ, (int)NQ, Pq, goutq, node, ba, bb, bc, big,
          fQ, fK, fV, wpack, apack, accb, mu, inv, stream);

  // --- Reciprocal attention (H=8, dk=dv=64) ---
  float* Qb  = big;                 // [NP,512]
  float* Vq  = big + NP * 512;      // [NP,512]
  float* o1h = big + 2 * NP * 512;  // [NP,512]
  float* Kb  = bb;                  // [NQ,512]
  float* Vp  = bb + NQ * 512;       // [NQ,512]
  float* o2h = bb + 2 * NQ * 512;   // [NQ,512]
  gemm(goutp, wq,  nullptr, nullptr, Qb, (int)NP, 512, 384, 0, wpack, apack, stream);
  gemm(goutq, wk,  nullptr, nullptr, Kb, (int)NQ, 512, 384, 0, wpack, apack, stream);
  gemm(goutq, wv,  nullptr, nullptr, Vp, (int)NQ, 512, 384, 0, wpack, apack, stream);
  gemm(goutp, wv2, nullptr, nullptr, Vq, (int)NP, 512, 384, 0, wpack, apack, stream);
  // f16 packs: Qf [NP,512] in fQ; Kf [NQ,512] in fK; Vp^T [512,NQ] after Kf;
  // Vq^T [512,NP] in fV.
  _Float16* Qf  = fQ;
  _Float16* Kf  = fK;
  _Float16* VpT = fK + NQ * 512;
  _Float16* VqT = fV;
  pack_kernel <<<cdiv256(NP * 512), 256, 0, stream>>>(Qb, (int)NP, 512, 512, 0, Qf);
  pack_kernel <<<cdiv256(NQ * 512), 256, 0, stream>>>(Kb, (int)NQ, 512, 512, 0, Kf);
  packT_kernel<<<cdiv256(NQ * 512), 256, 0, stream>>>(Vp, (int)NQ, 512, 512, 0, VpT);
  packT_kernel<<<cdiv256(NP * 512), 256, 0, stream>>>(Vq, (int)NP, 512, 512, 0, VqT);
  const float sc64 = 1.0f / sqrtf(64.0f);
  // a1: prot attends peptide (Q=Qf, K=Kf, V=Vp)
  flash_kernel<64><<<dim3(NP / 128, 8), 256, 0, stream>>>(
      Qf, 512, Kf, 512, VpT, (int)NQ, o1h, 512, 64, (int)NP, (int)NQ, sc64);
  // a2: softmax(s^T) == attention with Q=Kf, K=Qf, V=Vq
  flash_kernel<64><<<dim3(NQ / 128, 8), 256, 0, stream>>>(
      Kf, 512, Qf, 512, VqT, (int)NP, o2h, 512, 64, (int)NQ, (int)NP, sc64);
  gemm(o1h, fc1_w, fc1_b, nullptr, bc, (int)NP, 384, 512, 0, wpack, apack, stream);
  ln_res_kernel<<<(unsigned)NP, 256, 0, stream>>>(bc, goutp, ln1_g, ln1_b, node, 384);  // prot1 -> node
  gemm(o2h, fc2_w, fc2_b, nullptr, bc, (int)NQ, 384, 512, 0, wpack, apack, stream);
  ln_res_kernel<<<(unsigned)NQ, 256, 0, stream>>>(bc, goutq, ln2_g, ln2_b, ba, 384);    // pep1 -> ba

  // --- FFNs (write final LN straight into d_out) ---
  float* out_p = (float*)d_out;
  float* out_q = (float*)d_out + NP * DD;
  gemm(node, fpw1, fpb1, nullptr, big, (int)NP, 1536, 384, /*relu=*/1, wpack, apack, stream);
  gemm(big,  fpw2, fpb2, nullptr, bc,  (int)NP, 384, 1536, 0, wpack, apack, stream);
  ln_res_kernel<<<(unsigned)NP, 256, 0, stream>>>(bc, node, fpg, fpb, out_p, 384);
  gemm(ba,  fqw1, fqb1, nullptr, big, (int)NQ, 1536, 384, /*relu=*/1, wpack, apack, stream);
  gemm(big, fqw2, fqb2, nullptr, bc,  (int)NQ, 384, 1536, 0, wpack, apack, stream);
  ln_res_kernel<<<(unsigned)NQ, 256, 0, stream>>>(bc, ba, fqg, fqb, out_q, 384);
}
